// FP8Linear_79680233276228
// MI455X (gfx1250) — matrix operand order repaired and checked by reading, not compiled, not run on hardware
//
#include <hip/hip_runtime.h>

// ---------------------------------------------------------------------------
// FP8Linear for MI455X (gfx1250): C[M=16384, N=4096] = X[M,K] * W[N,K]^T * s
// v4: LDS double-buffered GEMM, BOTH operands staged as f16.
//   - Block 256 thr (8 waves, wave32), block tile 128x128, K-step 64.
//   - Each wave computes 64x32 via v_wmma_f32_16x16x32_f16 (acc = 64 VGPRs).
//   - Next K-tile loaded into registers BEFORE the compute phase (10
//     independent b128 loads/thread, pinned by an asm scheduling fence),
//     converted+stored to the alternate LDS buffer after compute.
//   - A: fp32 -> f16 once per block. B: e4m3 -> f16 once per block (exact:
//     bit-shift + *2^8). Fragment builds are pure ds_load_b128 + reg aliasing
//     -> conversion VALU off the WMMA critical path.
//   - LDS: As[2][128][72] + Bs[2][128][72] f16 = 72 KB (<= 320 KB/WGP on
//     CDNA5; 2 workgroups/WGP). 72-f16 stride = 36 dwords -> 16 lanes hit
//     16 disjoint 4-bank groups: conflict-free b128 access.
// ---------------------------------------------------------------------------

typedef __attribute__((ext_vector_type(16))) _Float16 v16h;
typedef __attribute__((ext_vector_type(8)))  _Float16 v8h;
typedef __attribute__((ext_vector_type(8)))  float    v8f;

#define K_DIM 4096
#define N_DIM 4096
#define BM 128
#define BN 128
#define KS 64
#define TSTR 72   // tile row stride in f16 (144B -> conflict-free b128)

// exact e4m3 -> f16: place exp/mant at f16 positions, then *2^8
// (fixes bias difference 15-7=8 for normals AND rescales denormals).
__device__ __forceinline__ _Float16 e4m3_to_h(unsigned int b) {
    unsigned short h = (unsigned short)(((b & 0x80u) << 8) | ((b & 0x7fu) << 7));
    _Float16 f = __builtin_bit_cast(_Float16, h);
    return f * (_Float16)256.0f;
}

// 8 raw fp8 bytes (two dwords) -> 8 f16 in linear order
__device__ __forceinline__ v8h cvt_b8(unsigned int w0, unsigned int w1) {
    v8h r;
    r[0] = e4m3_to_h(w0);       r[1] = e4m3_to_h(w0 >> 8);
    r[2] = e4m3_to_h(w0 >> 16); r[3] = e4m3_to_h(w0 >> 24);
    r[4] = e4m3_to_h(w1);       r[5] = e4m3_to_h(w1 >> 8);
    r[6] = e4m3_to_h(w1 >> 16); r[7] = e4m3_to_h(w1 >> 24);
    return r;
}

// 8 fp32 (two float4) -> 8 f16 in linear order
__device__ __forceinline__ v8h cvt8(float4 f0, float4 f1) {
    v8h r;
    r[0] = (_Float16)f0.x; r[1] = (_Float16)f0.y; r[2] = (_Float16)f0.z; r[3] = (_Float16)f0.w;
    r[4] = (_Float16)f1.x; r[5] = (_Float16)f1.y; r[6] = (_Float16)f1.z; r[7] = (_Float16)f1.w;
    return r;
}

// two v8h chunks from LDS -> one v16h fragment register block
__device__ __forceinline__ v16h frag2(v8h lo, v8h hi) {
    return __builtin_shufflevector(lo, hi,
        0, 1, 2, 3, 4, 5, 6, 7, 8, 9, 10, 11, 12, 13, 14, 15);
}

__global__ __launch_bounds__(256) void fp8linear_wmma_lds(
    const float* __restrict__ x,            // [M, K] fp32
    const unsigned char* __restrict__ w,    // [N, K] fp8 e4m3 bytes
    const float* __restrict__ scale,        // [1]
    float* __restrict__ out)                // [M, N] fp32
{
    __shared__ _Float16 As[2][BM][TSTR];
    __shared__ _Float16 Bs[2][BN][TSTR];

    const int tid  = threadIdx.x;
    const int lane = tid & 31;
    const int wave = tid >> 5;
    const int g = lane >> 4;          // half-wave group
    const int l = lane & 15;          // row/col within fragment
    const int wm = wave & 1;          // waves tiled 2(M) x 4(N)
    const int wn = wave >> 1;

    const int bm = blockIdx.y * BM;   // M panel (slow grid dim)
    const int bn = blockIdx.x * BN;   // N panel (fast grid dim -> x reuse in L2)
    const int m0w = wm * 64;          // wave origin within block tile
    const int n0w = wn * 32;

    const float s = scale[0];

    // ---- staging roles: thread t -> row t/2 (of 128), half t%2 (32 elems) ----
    const int srow  = tid >> 1;
    const int shalf = tid & 1;
    const float*         agp = x + (size_t)(bm + srow) * K_DIM + shalf * 32;
    const unsigned char* bgp = w + (size_t)(bn + srow) * K_DIM + shalf * 32;

    float4 ra[8];   // 32 fp32 of A per thread
    uint4  rb[2];   // 32 fp8 bytes of B per thread

    auto issue_loads = [&](int k) {
#pragma unroll
        for (int q = 0; q < 8; ++q) ra[q] = *(const float4*)(agp + k + 4 * q);
        rb[0] = *(const uint4*)(bgp + k);
        rb[1] = *(const uint4*)(bgp + k + 16);
        // scheduling fence: keep these loads issued before the compute phase
        asm volatile("" ::: "memory");
    };

    auto cvt_store = [&](int buf) {
        _Float16* ad = &As[buf][srow][shalf * 32];
#pragma unroll
        for (int q = 0; q < 4; ++q)
            *(v8h*)(ad + 8 * q) = cvt8(ra[2 * q], ra[2 * q + 1]);
        _Float16* bd = &Bs[buf][srow][shalf * 32];
        *(v8h*)(bd)      = cvt_b8(rb[0].x, rb[0].y);
        *(v8h*)(bd + 8)  = cvt_b8(rb[0].z, rb[0].w);
        *(v8h*)(bd + 16) = cvt_b8(rb[1].x, rb[1].y);
        *(v8h*)(bd + 24) = cvt_b8(rb[1].z, rb[1].w);
    };

    v8f acc[4][2] = {};

    issue_loads(0);
    cvt_store(0);
    __syncthreads();

    const int NK = K_DIM / KS;
    for (int kt = 0; kt < NK; ++kt) {
        const int  cur  = kt & 1;
        const bool more = (kt + 1) < NK;
        if (more) issue_loads((kt + 1) * KS);   // in flight during compute

        // ---- compute from LDS[cur]: 2 x (2 B-frags, 4 A-frags, 8 wmma) ----
#pragma unroll
        for (int kk = 0; kk < KS; kk += 32) {
            v16h bf[2];
#pragma unroll
            for (int j = 0; j < 2; ++j) {
                // B frag (32x16 KxN): lane = col N, elems = K kk+g*16 .. +15
                const _Float16* p = &Bs[cur][n0w + 16 * j + l][kk + g * 16];
                bf[j] = frag2(*(const v8h*)p, *(const v8h*)(p + 8));
            }
#pragma unroll
            for (int i = 0; i < 4; ++i) {
                // A frag (16x32 MxK): elems = K kk+g*16 .. +15
                const _Float16* p = &As[cur][m0w + 16 * i + l][kk + g * 16];
                v16h af = frag2(*(const v8h*)p, *(const v8h*)(p + 8));
#pragma unroll
                for (int j = 0; j < 2; ++j)
                    acc[i][j] = __builtin_amdgcn_wmma_f32_16x16x32_f16(
                        /*neg_a=*/false, af, /*neg_b=*/false, bf[j],
                        /*c_mod=*/(short)0, acc[i][j],
                        /*reuse_a=*/false, /*reuse_b=*/false);
            }
        }

        if (more) cvt_store((kt + 1) & 1);      // write the other buffer
        __syncthreads();
    }

    // ---- epilogue: C/D layout — VGPR r: lanes 0-15 row r, lanes 16-31 row r+8
#pragma unroll
    for (int i = 0; i < 4; ++i) {
#pragma unroll
        for (int j = 0; j < 2; ++j) {
            const int col = bn + n0w + 16 * j + l;
#pragma unroll
            for (int r = 0; r < 8; ++r) {
                const int row = bm + m0w + 16 * i + r + 8 * g;
                out[(size_t)row * N_DIM + col] = acc[i][j][r] * s;
            }
        }
    }
}

extern "C" void kernel_launch(void* const* d_in, const int* in_sizes, int n_in,
                              void* d_out, int out_size, void* d_ws, size_t ws_size,
                              hipStream_t stream) {
    const float*         xp = (const float*)d_in[0];         // [4,4096,4096] fp32
    const unsigned char* wp = (const unsigned char*)d_in[1]; // [4096,4096] e4m3 bytes
    const float*         sp = (const float*)d_in[2];         // [1]
    float*               op = (float*)d_out;                 // [4,4096,4096] fp32

    const int M = in_sizes[0] / K_DIM;     // 16384
    dim3 grid(N_DIM / BN, M / BM);         // (32, 128): N fastest for L2 reuse
    dim3 block(256);
    fp8linear_wmma_lds<<<grid, block, 0, stream>>>(xp, wp, sp, op);
}
